// GAT_69131793596603
// MI455X (gfx1250) — matrix operand order repaired and checked by reading, not compiled, not run on hardware
//
#include <hip/hip_runtime.h>

#define B_    2
#define N_    512
#define NODEH 256
#define EDGEH 128
#define H_    8
#define D_    32
#define HD_   256   // H*D

typedef __attribute__((ext_vector_type(16))) __bf16 v16bf;
typedef __attribute__((ext_vector_type(8)))  float  v8f;

// ---------- bf16 helpers (round-to-nearest-even) ----------
static __device__ __forceinline__ unsigned short f2bf(float f) {
  union { float f; unsigned u; } x; x.f = f;
  return (unsigned short)((x.u + 0x7fffu + ((x.u >> 16) & 1u)) >> 16);
}
static __device__ __forceinline__ unsigned packbf2(float a, float b) {
  return (unsigned)f2bf(a) | ((unsigned)f2bf(b) << 16);
}
static __device__ __forceinline__ v8f zero_v8f() {
  v8f z;
#pragma unroll
  for (int i = 0; i < 8; ++i) z[i] = 0.0f;
  return z;
}
static __device__ __forceinline__ v8f wmma_bf16(v16bf a, v16bf b, v8f c) {
  // D = A(16x32) * B(32x16) + C, f32 accumulate
  return __builtin_amdgcn_wmma_f32_16x16x32_bf16(false, a, false, b, (short)0, c,
                                                 false, false);
}

// ---------- CDNA5 async global->LDS copy (ISA 15.18.3, ASYNCcnt) ----------
// lds_off: LDS byte address (low 32 bits of the flat LDS pointer, ISA 10.2);
// gaddr:   64-bit global address (VGPR pair).
static __device__ __forceinline__ void async_g2l_b128(unsigned lds_off,
                                                      const void* gaddr) {
  asm volatile("global_load_async_to_lds_b128 %0, %1, off"
               :: "v"(lds_off), "v"(gaddr) : "memory");
}
static __device__ __forceinline__ void wait_asynccnt0() {
  asm volatile("s_wait_asynccnt 0x0" ::: "memory");
}

// A fragment (16x32 bf16) gathered from a row-major LDS tile (stride ldp halfs).
// M = lane&15; K offset +16 for lanes>=16; element e -> K = e (contiguous).
static __device__ __forceinline__ v16bf load_a(const unsigned short* lds, int ldp,
                                               int row0, int kbase, int lane) {
  union { v16bf v; unsigned u[8]; } f;
  const unsigned short* rp =
      lds + (row0 + (lane & 15)) * ldp + kbase + ((lane >> 4) << 4);
#pragma unroll
  for (int p = 0; p < 8; ++p)
    f.u[p] = *(const unsigned*)(rp + 2 * p);
  return f.v;
}

// B fragment (32x16 bf16) from fragment-packed global buffer:
// lane holds N = lane&15, K = (lane<16 ? 0 : 16) + e, stored contiguously (32B/lane).
static __device__ __forceinline__ v16bf load_b(const unsigned short* __restrict__ pk,
                                               int frag, int lane) {
  union { v16bf v; uint4 q[2]; } f;
  const unsigned short* p = pk + ((size_t)(frag * 32 + lane) << 4);
  f.q[0] = *(const uint4*)p;
  f.q[1] = *(const uint4*)(p + 8);
  return f.v;
}

// B fragment from LDS holding W transposed as [n][k] (stride wp halfs, 16B aligned).
static __device__ __forceinline__ v16bf load_b_t(const unsigned short* wt, int wp,
                                                 int n0, int kbase, int lane) {
  union { v16bf v; uint4 q[2]; } f;
  const unsigned short* p = wt + (n0 + (lane & 15)) * wp + kbase + ((lane >> 4) << 4);
  f.q[0] = *(const uint4*)p;
  f.q[1] = *(const uint4*)(p + 8);
  return f.v;
}

// ------------------------------------------------------------------
// Pack We [128,256] and Weo [256,128] (f32) into bf16 B-fragment layout.
// ------------------------------------------------------------------
__global__ void pack_weights(const float* __restrict__ We, const float* __restrict__ Weo,
                             unsigned short* __restrict__ Wepk,
                             unsigned short* __restrict__ Weopk) {
  int gid = blockIdx.x * 256 + threadIdx.x;
  if (gid < EDGEH * HD_) {                    // We: k in [0,128), n in [0,256)
    int k = gid >> 8, n = gid & 255;
    int kt = k >> 5, kl = k & 31, nt = n >> 4, nl = n & 15;
    int lane = nl + ((kl >> 4) << 4);
    int e = kl & 15;
    Wepk[((size_t)((kt * 16 + nt) * 32 + lane) << 4) + e] = f2bf(We[gid]);
  } else {                                    // Weo: k in [0,256), n in [0,128)
    int g = gid - EDGEH * HD_;
    int k = g >> 7, n = g & 127;
    int kt = k >> 5, kl = k & 31, nt = n >> 4, nl = n & 15;
    int lane = nl + ((kl >> 4) << 4);
    int e = kl & 15;
    Weopk[((size_t)((kt * 8 + nt) * 32 + lane) << 4) + e] = f2bf(Weo[g]);
  }
}

// ------------------------------------------------------------------
// C[M,Ncols] = A[M,256] @ W[256,Ncols] (+bias), bf16 WMMA, f32 in/out.
// Block: 256 threads (8 waves), tile 128 rows x 64 cols.
// ------------------------------------------------------------------
__global__ void gemm_k256(const float* __restrict__ A, const float* __restrict__ W,
                          const float* __restrict__ bias, float* __restrict__ C,
                          int Ncols) {
  extern __shared__ char smem[];
  const int AP = 260, WP = 264;
  unsigned short* a_lds = (unsigned short*)smem;                       // [128][260]
  unsigned short* wt_lds = (unsigned short*)(smem + 128 * AP * 2);     // [64][264] = W^T
  int t = threadIdx.x, lane = t & 31, wave = t >> 5;
  int rowBase = blockIdx.x * 128;
  int colBase = blockIdx.y * 64;

  { // stage A tile -> bf16 (each thread: half a row)
    int r = t >> 1, half = (t & 1) * 128;
    const float4* src = (const float4*)(A + (size_t)(rowBase + r) * 256 + half);
    unsigned short* dst = a_lds + r * AP + half;
#pragma unroll
    for (int c4 = 0; c4 < 32; ++c4) {
      float4 f = src[c4];
      *(uint2*)(dst + c4 * 4) = make_uint2(packbf2(f.x, f.y), packbf2(f.z, f.w));
    }
  }
  { // stage W panel transposed -> bf16 (thread t handles k = t)
    const float4* src = (const float4*)(W + (size_t)t * Ncols + colBase);
#pragma unroll
    for (int n4 = 0; n4 < 16; ++n4) {
      float4 f = src[n4];
      wt_lds[(n4 * 4 + 0) * WP + t] = f2bf(f.x);
      wt_lds[(n4 * 4 + 1) * WP + t] = f2bf(f.y);
      wt_lds[(n4 * 4 + 2) * WP + t] = f2bf(f.z);
      wt_lds[(n4 * 4 + 3) * WP + t] = f2bf(f.w);
    }
  }
  __syncthreads();

  int mt = wave;  // each wave: one 16-row tile, all 4 col tiles
  v8f acc[4];
#pragma unroll
  for (int nt = 0; nt < 4; ++nt) acc[nt] = zero_v8f();
#pragma unroll
  for (int kt = 0; kt < 8; ++kt) {
    v16bf a = load_a(a_lds, AP, mt * 16, kt * 32, lane);
#pragma unroll
    for (int nt = 0; nt < 4; ++nt) {
      v16bf b = load_b_t(wt_lds, WP, nt * 16, kt * 32, lane);
      acc[nt] = wmma_bf16(a, b, acc[nt]);
    }
  }
  int hi8 = (lane >> 4) << 3, ln = lane & 15;
#pragma unroll
  for (int nt = 0; nt < 4; ++nt) {
    int col = colBase + nt * 16 + ln;
    float bb = bias ? bias[col] : 0.0f;
#pragma unroll
    for (int r = 0; r < 8; ++r) {
      int row = rowBase + mt * 16 + r + hi8;
      C[(size_t)row * Ncols + col] = acc[nt][r] + bb;
    }
  }
}

// ------------------------------------------------------------------
// Fused edge kernel: per 16x16 (i,j) tile:
//   e   = edge_tile @ We                      (bf16 WMMA, f32 acc)
//   raw = e + q_i * k_j                       (f32, then bf16 to LDS)
//   s[b,i,j,h] = sum_d raw                    (cross-lane reduce)
//   edge_out = raw @ Weo + beo                (bf16 WMMA)
// raw tile reuses the edge tile's LDS region (rows are wave-private);
// split into two K-halves to keep LDS at ~108KB (CU-mode safe).
// q/k tiles are staged with async global->LDS copies overlapped with the
// VALU-heavy f32->bf16 edge-tile conversion.
// ------------------------------------------------------------------
__global__ void fused_edge(const float* __restrict__ edge_attr,
                           const float* __restrict__ qk,
                           const unsigned short* __restrict__ Wepk,
                           const unsigned short* __restrict__ Weopk,
                           const float* __restrict__ beo,
                           float* __restrict__ edge_out,
                           float* __restrict__ s_ws) {
  extern __shared__ char smem[];
  const int EP = 132;  // stride (halfs) for both edge tile and raw half-tile
  unsigned short* e_lds = (unsigned short*)smem;      // [256][132] (reused as raw)
  float* q_lds = (float*)(smem + 256 * EP * 2);       // [16][256]
  float* k_lds = q_lds + 16 * 256;                    // [16][256]
  float* s_tile = k_lds + 16 * 256;                   // [256][8]

  int t = threadIdx.x, lane = t & 31, wave = t >> 5;
  int blk = blockIdx.x;
  int b = blk >> 10;
  int it = (blk >> 5) & 31, jt = blk & 31;
  int i0 = it * 16, j0 = jt * 16;

  { // async-stage q rows (i-tile) and k rows (j-tile), f32, 16B chunks
    unsigned qoff = (unsigned)(size_t)q_lds;   // LDS byte address (low 32 bits)
    unsigned koff = (unsigned)(size_t)k_lds;
#pragma unroll
    for (int c = 0; c < 4; ++c) {
      int chunk = t + c * 256;                 // 0..1023, 16B each (16 rows x 1KB)
      int r16 = chunk >> 6, c64 = chunk & 63;
      async_g2l_b128(qoff + chunk * 16,
                     qk + ((size_t)b * N_ + (i0 + r16)) * 512 + c64 * 4);
      async_g2l_b128(koff + chunk * 16,
                     qk + ((size_t)b * N_ + (j0 + r16)) * 512 + 256 + c64 * 4);
    }
  }
  { // stage edge tile (row t = (di,dj) pair) as bf16
    int di = t >> 4, dj = t & 15;
    const float4* src = (const float4*)(edge_attr +
        ((((size_t)b * N_) + (i0 + di)) * N_ + (j0 + dj)) * EDGEH);
    unsigned short* dst = e_lds + t * EP;
#pragma unroll
    for (int c4 = 0; c4 < 32; ++c4) {
      float4 f = src[c4];
      *(uint2*)(dst + c4 * 4) = make_uint2(packbf2(f.x, f.y), packbf2(f.z, f.w));
    }
  }
  wait_asynccnt0();
  __syncthreads();

  int hi8 = (lane >> 4) << 3, ln = lane & 15;

  // Preload this wave's edge A-fragments (rows 32*wave .. +31) before overwrite.
  v16bf afr[2][4];
#pragma unroll
  for (int m = 0; m < 2; ++m)
#pragma unroll
    for (int kt = 0; kt < 4; ++kt)
      afr[m][kt] = load_a(e_lds, EP, (wave * 2 + m) * 16, kt * 32, lane);

  v8f acc2[2][8];
#pragma unroll
  for (int m = 0; m < 2; ++m)
#pragma unroll
    for (int nt = 0; nt < 8; ++nt) acc2[m][nt] = zero_v8f();

  for (int half = 0; half < 2; ++half) {
    // ---- phase 1: e-GEMM + q*k + head sums for channel cols [half*128, half*128+128)
    for (int hh = 0; hh < 4; ++hh) {
      int h = half * 4 + hh;
      float hsum[2][8];
#pragma unroll
      for (int m = 0; m < 2; ++m)
#pragma unroll
        for (int r = 0; r < 8; ++r) hsum[m][r] = 0.0f;
#pragma unroll
      for (int sub = 0; sub < 2; ++sub) {
        int ct = h * 2 + sub;
        v16bf bfr[4];
#pragma unroll
        for (int kt = 0; kt < 4; ++kt) bfr[kt] = load_b(Wepk, kt * 16 + ct, lane);
        int c = ct * 16 + ln;
#pragma unroll
        for (int m = 0; m < 2; ++m) {
          int mt = wave * 2 + m;
          v8f acc = zero_v8f();
#pragma unroll
          for (int kt = 0; kt < 4; ++kt) acc = wmma_bf16(afr[m][kt], bfr[kt], acc);
          float qv = q_lds[mt * 256 + c];
#pragma unroll
          for (int r = 0; r < 8; ++r) {
            float val = acc[r] + qv * k_lds[(r + hi8) * 256 + c];
            e_lds[(mt * 16 + r + hi8) * EP + (c & 127)] = f2bf(val);  // raw, bf16
            hsum[m][r] += val;
          }
        }
      }
      // reduce over the 16 columns held across lanes (halves stay separate)
#pragma unroll
      for (int m = 0; m < 2; ++m)
#pragma unroll
        for (int r = 0; r < 8; ++r) {
          float v = hsum[m][r];
          v += __shfl_xor(v, 1, 32);
          v += __shfl_xor(v, 2, 32);
          v += __shfl_xor(v, 4, 32);
          v += __shfl_xor(v, 8, 32);
          if (ln == 0)
            s_tile[((wave * 2 + m) * 16 + r + hi8) * 8 + h] = v;
        }
    }
    // ---- phase 2 (partial K): edge_out accumulation; B-fragments shared by both
    // row tiles to halve Weo L2 traffic.
    v16bf ar2[2][4];
#pragma unroll
    for (int m = 0; m < 2; ++m)
#pragma unroll
      for (int k2 = 0; k2 < 4; ++k2)
        ar2[m][k2] = load_a(e_lds, EP, (wave * 2 + m) * 16, k2 * 32, lane);
#pragma unroll
    for (int nt = 0; nt < 8; ++nt)
#pragma unroll
      for (int k2 = 0; k2 < 4; ++k2) {
        v16bf bb = load_b(Weopk, (half * 4 + k2) * 8 + nt, lane);
        acc2[0][nt] = wmma_bf16(ar2[0][k2], bb, acc2[0][nt]);
        acc2[1][nt] = wmma_bf16(ar2[1][k2], bb, acc2[1][nt]);
      }
  }
  __syncthreads();

  { // write attention-score tile to global s[b,i,j,h]
    int di = t >> 4, dj = t & 15;
    float4* dst = (float4*)(s_ws +
        ((((size_t)b * N_) + (i0 + di)) * N_ + (j0 + dj)) * H_);
    const float4* sp = (const float4*)(s_tile + t * 8);
    dst[0] = sp[0];
    dst[1] = sp[1];
  }

  // edge_out epilogue
#pragma unroll
  for (int m = 0; m < 2; ++m) {
    int gi = i0 + wave * 2 + m;
#pragma unroll
    for (int nt = 0; nt < 8; ++nt) {
      int col = nt * 16 + ln;
      float bo = beo[col];
#pragma unroll
      for (int r = 0; r < 8; ++r) {
        int gj = j0 + r + hi8;
        edge_out[((((size_t)b * N_) + gi) * N_ + gj) * EDGEH + col] = acc2[m][nt][r] + bo;
      }
    }
  }
}

// ------------------------------------------------------------------
// Softmax over j (masked, scaled) + node_hidden[b,i,h,d] = sum_j att * v
// One block per (b,i); wave w handles head w; lane = d.
// ------------------------------------------------------------------
__global__ void softmax_av(const float* __restrict__ s_ws,
                           const float* __restrict__ v_ws,
                           const unsigned char* __restrict__ mask,
                           float* __restrict__ hidden) {
  extern __shared__ char smem[];
  float* att = (float*)smem;  // [8][512]
  int t = threadIdx.x, lane = t & 31, wave = t >> 5;
  int bi = blockIdx.x;        // b*N + i
  int b = bi >> 9;
  const float scale = 0.17677669529663687f;  // 1/sqrt(D)
  size_t sbase = (size_t)bi * N_ * H_;
  for (int idx = t; idx < N_ * H_; idx += 256) {
    int j = idx >> 3, h = idx & 7;
    float v = s_ws[sbase + idx] * scale;
    if (!mask[b * N_ + j]) v = -3.0e38f;
    att[h * N_ + j] = v;
  }
  __syncthreads();

  int h = wave;
  float m = -3.0e38f;
#pragma unroll
  for (int jj = 0; jj < 16; ++jj) m = fmaxf(m, att[h * N_ + lane + jj * 32]);
  for (int off = 16; off > 0; off >>= 1) m = fmaxf(m, __shfl_xor(m, off, 32));
  float sum = 0.0f, ex[16];
#pragma unroll
  for (int jj = 0; jj < 16; ++jj) {
    ex[jj] = __expf(att[h * N_ + lane + jj * 32] - m);
    sum += ex[jj];
  }
  for (int off = 16; off > 0; off >>= 1) sum += __shfl_xor(sum, off, 32);
  float inv = 1.0f / sum;
#pragma unroll
  for (int jj = 0; jj < 16; ++jj) att[h * N_ + lane + jj * 32] = ex[jj] * inv;
  // per-wave LDS ops are in-order; only this wave reads att[h][*] below.
  float o = 0.0f;
  for (int j = 0; j < N_; ++j)
    o += att[h * N_ + j] * v_ws[((size_t)b * N_ + j) * HD_ + h * D_ + lane];
  hidden[(size_t)bi * HD_ + h * D_ + lane] = o;
}

// ------------------------------------------------------------------
extern "C" void kernel_launch(void* const* d_in, const int* in_sizes, int n_in,
                              void* d_out, int out_size, void* d_ws, size_t ws_size,
                              hipStream_t stream) {
  (void)in_sizes; (void)n_in; (void)out_size; (void)ws_size;
  const float* x         = (const float*)d_in[0];
  const float* edge_attr = (const float*)d_in[1];
  const unsigned char* mask = (const unsigned char*)d_in[2];
  const float* Wqk = (const float*)d_in[3];
  const float* We  = (const float*)d_in[4];
  const float* Wv  = (const float*)d_in[5];
  const float* bv  = (const float*)d_in[6];
  const float* Wno = (const float*)d_in[7];
  const float* bno = (const float*)d_in[8];
  const float* Weo = (const float*)d_in[9];
  const float* beo = (const float*)d_in[10];

  float* node_out = (float*)d_out;
  float* edge_out = node_out + (size_t)B_ * N_ * NODEH;

  char* ws = (char*)d_ws;
  size_t off = 0;
  float* qk_ws = (float*)(ws + off); off += (size_t)B_ * N_ * 512 * 4;
  float* v_ws  = (float*)(ws + off); off += (size_t)B_ * N_ * 256 * 4;
  float* hid_ws= (float*)(ws + off); off += (size_t)B_ * N_ * 256 * 4;
  float* s_ws  = (float*)(ws + off); off += (size_t)B_ * N_ * N_ * H_ * 4;
  unsigned short* Wepk  = (unsigned short*)(ws + off); off += 32768 * 2;
  unsigned short* Weopk = (unsigned short*)(ws + off); off += 32768 * 2;

  pack_weights<<<256, 256, 0, stream>>>(We, Weo, Wepk, Weopk);

  size_t gemm_lds = 128 * 260 * 2 + 64 * 264 * 2;   // ~100 KB
  gemm_k256<<<dim3(8, 8), 256, gemm_lds, stream>>>(x, Wqk, nullptr, qk_ws, 512);
  gemm_k256<<<dim3(8, 4), 256, gemm_lds, stream>>>(x, Wv, bv, v_ws, 256);

  size_t fused_lds = 256 * 132 * 2 + 16 * 256 * 4 * 2 + 256 * 8 * 4;  // ~108 KB
  fused_edge<<<2048, 256, fused_lds, stream>>>(edge_attr, qk_ws, Wepk, Weopk, beo,
                                               edge_out, s_ws);

  softmax_av<<<B_ * N_, 256, (size_t)H_ * N_ * 4, stream>>>(s_ws, v_ws, mask, hid_ws);

  gemm_k256<<<dim3(8, 4), 256, gemm_lds, stream>>>(hid_ws, Wno, bno, node_out, 256);
}
